// UnsupervisedGIN_38113539785114
// MI455X (gfx1250) — matrix operand order repaired and checked by reading, not compiled, not run on hardware
//
#include <hip/hip_runtime.h>
#include <hip/hip_bf16.h>

#define N_NODES   16384
#define N_EDGES   262144
#define N_GRAPHS  64
#define IN_DIM    64
#define HID       512
#define BN_EPS    1e-5f

typedef __attribute__((ext_vector_type(16))) __bf16 v16bf;
typedef __attribute__((ext_vector_type(8)))  __bf16 v8bf;
typedef __attribute__((ext_vector_type(8)))  float  v8f;

#define SHUF16(lo, hi) __builtin_shufflevector(lo, hi, \
    0, 1, 2, 3, 4, 5, 6, 7, 8, 9, 10, 11, 12, 13, 14, 15)

// ---------------- utility ----------------
__global__ void k_zero(float* __restrict__ p, int n) {
    int i = blockIdx.x * blockDim.x + threadIdx.x;
    if (i < n) p[i] = 0.0f;
}

__global__ void k_deg(const int* __restrict__ dst, float* __restrict__ deg) {
    int e = blockIdx.x * blockDim.x + threadIdx.x;
    if (e < N_EDGES) atomicAdd(&deg[dst[e]], 1.0f);
}

// gather x[src] -> atomic scatter-add to AGG[dst], float4 vectorized
__global__ void k_scatter4(const float* __restrict__ X, const int* __restrict__ src,
                           const int* __restrict__ dst, float* __restrict__ AGG, int d) {
    int tid = blockIdx.x * blockDim.x + threadIdx.x;
    int cpe = d >> 2;                       // float4 chunks per edge
    if (tid >= N_EDGES * cpe) return;
    int e = tid / cpe;
    int c = (tid - e * cpe) << 2;
    const float4 v = *(const float4*)(X + (size_t)src[e] * d + c);
    float* o = AGG + (size_t)dst[e] * d + c;
    atomicAdd(o + 0, v.x); atomicAdd(o + 1, v.y);
    atomicAdd(o + 2, v.z); atomicAdd(o + 3, v.w);
}

// h = x + agg/deg, packed to bf16 for WMMA
__global__ void k_h_bf16(const float* __restrict__ X, const float* __restrict__ AGG,
                         const float* __restrict__ deg, __bf16* __restrict__ Hb, int d) {
    int tid = blockIdx.x * blockDim.x + threadIdx.x;
    if (tid >= N_NODES * d) return;
    int i = tid / d;
    float dv = deg[i]; dv = dv < 1.0f ? 1.0f : dv;
    Hb[tid] = (__bf16)(X[tid] + AGG[tid] / dv);
}

__global__ void k_convw(const float* __restrict__ W, __bf16* __restrict__ Wb, int n) {
    int i = blockIdx.x * blockDim.x + threadIdx.x;
    if (i < n) Wb[i] = (__bf16)W[i];
}

// ---------------- bf16 WMMA GEMM: C[M=16384,N=512] = A[M,K] @ B[K,512] + bias ----------------
// Each wave owns a 32x32 output block = 2x2 WMMA tiles (4 f32 accumulators).
// Per K-step(32): 2 A fragments + 2 B fragments feed 4 v_wmma_f32_16x16x32_bf16,
// halving L1 operand traffic per matrix op vs 1 tile/wave.
__global__ __launch_bounds__(256) void k_gemm(const __bf16* __restrict__ A,
                                              const __bf16* __restrict__ B,
                                              const float* __restrict__ bias,
                                              float* __restrict__ C, int K) {
    const int lane = threadIdx.x & 31;
    const int wave = threadIdx.x >> 5;
    const int job  = blockIdx.x * 8 + wave;     // (M/32)*(N/32) jobs, 8 consecutive N-blocks/block
    const int NJ   = HID / 32;                  // 16
    const int m0   = (job / NJ) << 5;
    const int n0   = (job % NJ) << 5;
    const int half = lane >> 4;                 // 0: lanes 0-15, 1: lanes 16-31
    const int r16  = lane & 15;
    const int koff = half * 8;                  // A K-group offset within 32

    v8f c00, c01, c10, c11;
    float bv0 = bias[n0 + r16];
    float bv1 = bias[n0 + 16 + r16];
#pragma unroll
    for (int r = 0; r < 8; ++r) { c00[r] = bv0; c01[r] = bv1; c10[r] = bv0; c11[r] = bv1; }

    const __bf16* arow0 = A + (size_t)(m0 + r16) * K;        // rows m0..m0+15
    const __bf16* arow1 = arow0 + (size_t)16 * K;            // rows m0+16..m0+31
    for (int k0 = 0; k0 < K; k0 += 32) {
        // A 16x32 bf16 fragments: lane holds its row, K = koff..+7 and koff+16..+23
        v8bf a0lo = *(const v8bf*)(arow0 + k0 + koff);
        v8bf a0hi = *(const v8bf*)(arow0 + k0 + koff + 16);
        v8bf a1lo = *(const v8bf*)(arow1 + k0 + koff);
        v8bf a1hi = *(const v8bf*)(arow1 + k0 + koff + 16);
        v16bf a0 = SHUF16(a0lo, a0hi);
        v16bf a1 = SHUF16(a1lo, a1hi);
        // B 32x16 bf16 fragments: lane holds its column, K = koff..+7 and koff+16..+23
        const __bf16* brow = B + (size_t)(k0 + koff) * HID + n0 + r16;
        v16bf b0, b1;
#pragma unroll
        for (int i = 0; i < 8; ++i) {
            b0[i]     = brow[(size_t)i * HID];
            b0[i + 8] = brow[(size_t)(i + 16) * HID];
            b1[i]     = brow[(size_t)i * HID + 16];
            b1[i + 8] = brow[(size_t)(i + 16) * HID + 16];
        }

        c00 = __builtin_amdgcn_wmma_f32_16x16x32_bf16(false, a0, false, b0, (short)0, c00, false, false);
        c01 = __builtin_amdgcn_wmma_f32_16x16x32_bf16(false, a0, false, b1, (short)0, c01, false, false);
        c10 = __builtin_amdgcn_wmma_f32_16x16x32_bf16(false, a1, false, b0, (short)0, c10, false, false);
        c11 = __builtin_amdgcn_wmma_f32_16x16x32_bf16(false, a1, false, b1, (short)0, c11, false, false);
    }

    // D layout: VGPR r -> row (tile_m0 + r + 8*half), col = tile_n0 + (lane&15)
    float* p0 = C + (size_t)(m0 + 8 * half) * HID + n0 + r16;       // tiles (m0, *)
    float* p1 = C + (size_t)(m0 + 16 + 8 * half) * HID + n0 + r16;  // tiles (m0+16, *)
#pragma unroll
    for (int r = 0; r < 8; ++r) {
        p0[(size_t)r * HID]      = c00[r];
        p0[(size_t)r * HID + 16] = c01[r];
        p1[(size_t)r * HID]      = c10[r];
        p1[(size_t)r * HID + 16] = c11[r];
    }
}

// ---------------- BatchNorm column stats (training mode, biased var) ----------------
__global__ void k_stats(const float* __restrict__ Y, const float* __restrict__ g,
                        const float* __restrict__ be, float* __restrict__ scale,
                        float* __restrict__ shift) {
    __shared__ float s1[256], s2[256];
    const int j = blockIdx.x;
    const int t = threadIdx.x;
    float sum = 0.0f, sq = 0.0f;
    for (int r = t; r < N_NODES; r += 256) {
        float v = Y[(size_t)r * HID + j];
        sum += v; sq += v * v;
    }
    s1[t] = sum; s2[t] = sq;
    __syncthreads();
    for (int s = 128; s > 0; s >>= 1) {
        if (t < s) { s1[t] += s1[t + s]; s2[t] += s2[t + s]; }
        __syncthreads();
    }
    if (t == 0) {
        float mean = s1[0] * (1.0f / N_NODES);
        float var  = s2[0] * (1.0f / N_NODES) - mean * mean;
        float rstd = rsqrtf(var + BN_EPS);
        float sc   = g[j] * rstd;
        scale[j] = sc;
        shift[j] = be[j] - mean * sc;
    }
}

__global__ void k_bnrelu_bf16(const float* __restrict__ Y, const float* __restrict__ scale,
                              const float* __restrict__ shift, __bf16* __restrict__ O) {
    int tid = blockIdx.x * blockDim.x + threadIdx.x;
    if (tid >= N_NODES * HID) return;
    int j = tid & (HID - 1);
    float v = fmaf(Y[tid], scale[j], shift[j]);
    O[tid] = (__bf16)(v > 0.0f ? v : 0.0f);
}

__global__ void k_bnrelu_f32(const float* __restrict__ Y, const float* __restrict__ scale,
                             const float* __restrict__ shift, float* __restrict__ O) {
    int tid = blockIdx.x * blockDim.x + threadIdx.x;
    if (tid >= N_NODES * HID) return;
    int j = tid & (HID - 1);
    float v = fmaf(Y[tid], scale[j], shift[j]);
    O[tid] = v > 0.0f ? v : 0.0f;
}

// ---------------- readout ----------------
__global__ void k_cnt(const int* __restrict__ gid, float* __restrict__ cnt) {
    int i = blockIdx.x * blockDim.x + threadIdx.x;
    if (i < N_NODES) atomicAdd(&cnt[gid[i]], 1.0f);
}

__global__ void k_pool(const float* __restrict__ X, const int* __restrict__ gid,
                       float* __restrict__ pool) {
    int tid = blockIdx.x * blockDim.x + threadIdx.x;
    const int cpe = HID >> 2;
    if (tid >= N_NODES * cpe) return;
    int i = tid / cpe;
    int c = (tid - i * cpe) << 2;
    float4 v = *(const float4*)(X + (size_t)i * HID + c);
    float* o = pool + (size_t)gid[i] * HID + c;
    atomicAdd(o + 0, v.x); atomicAdd(o + 1, v.y);
    atomicAdd(o + 2, v.z); atomicAdd(o + 3, v.w);
}

__global__ void k_pooldiv(const float* __restrict__ pool, const float* __restrict__ cnt,
                          float* __restrict__ out) {
    int tid = blockIdx.x * blockDim.x + threadIdx.x;
    if (tid >= N_GRAPHS * HID) return;
    int g = tid / HID;
    float c = cnt[g]; c = c < 1.0f ? 1.0f : c;
    out[tid] = pool[tid] / c;
}

// ---------------- orchestration ----------------
static inline int cdiv(long long a, int b) { return (int)((a + b - 1) / b); }

extern "C" void kernel_launch(void* const* d_in, const int* in_sizes, int n_in,
                              void* d_out, int out_size, void* d_ws, size_t ws_size,
                              hipStream_t stream) {
    (void)in_sizes; (void)n_in; (void)out_size; (void)ws_size;
    const float* features = (const float*)d_in[0];
    const int*   src      = (const int*)d_in[1];
    const int*   dst      = (const int*)d_in[2];
    const int*   gid      = (const int*)d_in[3];
    // d_in[4] = num_graphs (compile-time 64 here)
    const float* P[16];
    for (int i = 0; i < 16; ++i) P[i] = (const float*)d_in[5 + i];
    // per layer l (l=0,1): P[8l+0]=W1, +1=b1, +2=g1, +3=be1, +4=W2, +5=b2, +6=g2, +7=be2

    float* out    = (float*)d_out;
    float* Xact   = out + (size_t)N_GRAPHS * HID;   // x region of d_out doubles as activation buf

    char* w = (char*)d_ws;
    float*  deg   = (float*)w;  w += (size_t)N_NODES * 4;
    float*  F     = (float*)w;  w += (size_t)N_NODES * HID * 4;   // AGG and GEMM output share
    __bf16* Hb    = (__bf16*)w; w += (size_t)N_NODES * HID * 2;
    __bf16* Wb    = (__bf16*)w; w += (size_t)HID * HID * 2;
    float*  scale = (float*)w;  w += HID * 4;
    float*  shift = (float*)w;  w += HID * 4;
    float*  pool  = (float*)w;  w += (size_t)N_GRAPHS * HID * 4;
    float*  cnt   = (float*)w;  w += N_GRAPHS * 4;

    const int TPB = 256;
    const int gemm_blocks = (N_NODES / 32) * (HID / 32) / 8;   // 1024 blocks of 8 wave-jobs

    // in-degree (shared by both layers)
    k_zero<<<cdiv(N_NODES, TPB), TPB, 0, stream>>>(deg, N_NODES);
    k_deg<<<cdiv(N_EDGES, TPB), TPB, 0, stream>>>(dst, deg);

    for (int l = 0; l < 2; ++l) {
        const int d = (l == 0) ? IN_DIM : HID;
        const float* Xin = (l == 0) ? features : Xact;
        const float* const* Q = &P[8 * l];

        // GIN aggregation: agg = segment_sum(x[src], dst) / deg ; h = x + agg
        k_zero<<<cdiv((long long)N_NODES * d, TPB), TPB, 0, stream>>>(F, N_NODES * d);
        k_scatter4<<<cdiv((long long)N_EDGES * (d / 4), TPB), TPB, 0, stream>>>(Xin, src, dst, F, d);
        k_h_bf16<<<cdiv((long long)N_NODES * d, TPB), TPB, 0, stream>>>(Xin, F, deg, Hb, d);

        // Linear1 + BN1 + ReLU
        k_convw<<<cdiv(d * HID, TPB), TPB, 0, stream>>>(Q[0], Wb, d * HID);
        k_gemm<<<gemm_blocks, TPB, 0, stream>>>(Hb, Wb, Q[1], F, d);
        k_stats<<<HID, TPB, 0, stream>>>(F, Q[2], Q[3], scale, shift);
        k_bnrelu_bf16<<<cdiv((long long)N_NODES * HID, TPB), TPB, 0, stream>>>(F, scale, shift, Hb);

        // Linear2 + BN2 + ReLU -> f32 activations (layer output)
        k_convw<<<cdiv(HID * HID, TPB), TPB, 0, stream>>>(Q[4], Wb, HID * HID);
        k_gemm<<<gemm_blocks, TPB, 0, stream>>>(Hb, Wb, Q[5], F, HID);
        k_stats<<<HID, TPB, 0, stream>>>(F, Q[6], Q[7], scale, shift);
        k_bnrelu_f32<<<cdiv((long long)N_NODES * HID, TPB), TPB, 0, stream>>>(F, scale, shift, Xact);
    }

    // AvgPooling readout per graph
    k_zero<<<cdiv(N_GRAPHS * HID, TPB), TPB, 0, stream>>>(pool, N_GRAPHS * HID);
    k_zero<<<1, TPB, 0, stream>>>(cnt, N_GRAPHS);
    k_cnt<<<cdiv(N_NODES, TPB), TPB, 0, stream>>>(gid, cnt);
    k_pool<<<cdiv((long long)N_NODES * (HID / 4), TPB), TPB, 0, stream>>>(Xact, gid, pool);
    k_pooldiv<<<cdiv(N_GRAPHS * HID, TPB), TPB, 0, stream>>>(pool, cnt, out);
}
